// SwinTransformerBlock_30923764531906
// MI455X (gfx1250) — matrix operand order repaired and checked by reading, not compiled, not run on hardware
//
#include <hip/hip_runtime.h>
#include <hip/hip_bf16.h>

typedef __bf16 v16bf __attribute__((ext_vector_type(16)));
typedef float  v8f   __attribute__((ext_vector_type(8)));

#define SHIFT 3

struct Frag32B { uint4 lo, hi; };   // 32 bytes, bit-castable to v16bf

__device__ __host__ inline __bf16 f2bf(float f) {
    union { float f; unsigned u; } a; a.f = f;
    unsigned u = a.u;
    unsigned r = (u + 0x7FFFu + ((u >> 16) & 1u)) >> 16;
    union { unsigned short s; __bf16 b; } o; o.s = (unsigned short)r;
    return o.b;
}

// ---- WMMA fragment loaders (CDNA5 16-bit layouts, ISA 7.12.2), b128-vectorized ----
// A (16x32): lane m=lane&15, g=lane>>4; elems 0..7 -> K=8g+e, elems 8..15 -> K=16+8g+e.
// Per lane: chunk0 = 16B at byte offset 16*g, chunk1 = 16B at byte offset 32+16*g
// (row base + kbase is always 32B aligned here).
__device__ inline v16bf load_a(const __bf16* base, int ld, int m0, int kbase, int lane) {
    const int g = lane >> 4, m = lane & 15;
    const uint4* p = (const uint4*)(base + (size_t)(m0 + m) * ld + kbase);
    Frag32B f; f.lo = p[g]; f.hi = p[2 + g];
    return __builtin_bit_cast(v16bf, f);
}
// B (32x16) from K-contiguous (transposed) storage: lane n=lane&15, g=lane>>4; elems 0..7 -> K=8g+e, elems 8..15 -> K=16+8g+e.
// Per lane: chunk0 = 16B at byte offset 16*g, chunk1 = 16B at byte offset 32+16*g.
__device__ inline v16bf load_b(const __bf16* base, int ld, int n0, int kbase, int lane) {
    const int g = lane >> 4, n = lane & 15;
    const uint4* p = (const uint4*)(base + (size_t)(n0 + n) * ld + kbase);
    Frag32B f; f.lo = p[g]; f.hi = p[2 + g];
    return __builtin_bit_cast(v16bf, f);
}
__device__ inline v8f wmma_bf16(v16bf a, v16bf b, v8f c) {
    return __builtin_amdgcn_wmma_f32_16x16x32_bf16(false, a, false, b, (short)0, c, false, false);
}

// =================== weight convert + transpose to bf16 ===================
__global__ __launch_bounds__(256) void swin_wconv(
    const float* __restrict__ qw, const float* __restrict__ kw,
    const float* __restrict__ vw, const float* __restrict__ pw,
    const float* __restrict__ f1, const float* __restrict__ f2,
    __bf16* __restrict__ ws)
{
    __bf16* qT  = ws;                 // 256 x 512
    __bf16* kT  = qT  + 256 * 512;    // 256 x 512
    __bf16* vT  = kT  + 256 * 512;    // 256 x 256
    __bf16* pT  = vT  + 256 * 256;    // 256 x 256
    __bf16* f1T = pT  + 256 * 256;    // 1024 x 256
    __bf16* f2T = f1T + 1024 * 256;   // 256 x 1024
    int i = blockIdx.x * 256 + threadIdx.x;
    if (i < 131072) { int n = i >> 9, k = i & 511; qT[i] = f2bf(qw[k * 256 + n]); return; }
    i -= 131072;
    if (i < 131072) { int n = i >> 9, k = i & 511; kT[i] = f2bf(kw[k * 256 + n]); return; }
    i -= 131072;
    if (i < 65536)  { int n = i >> 8, k = i & 255; vT[i] = f2bf(vw[k * 256 + n]); return; }
    i -= 65536;
    if (i < 65536)  { int n = i >> 8, k = i & 255; pT[i] = f2bf(pw[k * 256 + n]); return; }
    i -= 65536;
    if (i < 262144) { int n = i >> 8, k = i & 255; f1T[i] = f2bf(f1[k * 1024 + n]); return; }
    i -= 262144;
    if (i < 262144) { int n = i >> 10, k = i & 1023; f2T[i] = f2bf(f2[k * 256 + n]); return; }
}

// =================== fused LN1 + window attention + proj + residual ===================
__global__ __launch_bounds__(256) void swin_attn(
    const float* __restrict__ x, const float* __restrict__ guid,
    const float* __restrict__ n1w, const float* __restrict__ n1b,
    const __bf16* __restrict__ qT, const float* __restrict__ qb,
    const __bf16* __restrict__ kT, const float* __restrict__ kb,
    const __bf16* __restrict__ vT, const float* __restrict__ vb,
    const __bf16* __restrict__ pT, const float* __restrict__ pb,
    float* __restrict__ x2out)
{
    extern __shared__ char smem[];
    __bf16* Qs  = (__bf16*)smem;              // 64x256 bf16
    __bf16* Ks  = Qs + 64 * 256;              // 64x256
    __bf16* Vt  = Ks + 64 * 256;              // 256x64 (channel-major)
    float*  Ss  = (float*)(Vt + 64 * 256);    // 8 x 64x64 f32 (win overlays this)
    __bf16* Ws  = (__bf16*)Ss;                // 64x512 bf16 (phase-1 only)
    __bf16* Os  = (__bf16*)((char*)Ss + 8 * 4096 * 4); // 64x256
    int* pixs   = (int*)(Os + 64 * 256);
    int* rids   = pixs + 64;

    const int tid = threadIdx.x, lane = tid & 31, wave = tid >> 5;
    const int bidx = blockIdx.x;
    const int b = bidx >> 6, wi = bidx & 63, wh = wi >> 3, ww = wi & 7;

    // ---- phase 1: LN1 + concat + roll-gather into Ws; record pixel + region id ----
    for (int t = wave; t < 64; t += 8) {
        if (t < 49) {
            int th = t / 7, tw = t % 7;
            int hs = (wh * 7 + th + SHIFT) % 56;
            int wsx = (ww * 7 + tw + SHIFT) % 56;
            int pix = hs * 56 + wsx;
            const float* xr = x    + ((size_t)b * 3136 + pix) * 256;
            const float* gr = guid + ((size_t)b * 3136 + pix) * 256;
            float v[8]; float s = 0.f;
#pragma unroll
            for (int j = 0; j < 8; ++j) { v[j] = xr[j * 32 + lane]; s += v[j]; }
            for (int m = 16; m >= 1; m >>= 1) s += __shfl_xor(s, m, 32);
            float mu = s * (1.f / 256.f), vs = 0.f;
#pragma unroll
            for (int j = 0; j < 8; ++j) { float d = v[j] - mu; vs += d * d; }
            for (int m = 16; m >= 1; m >>= 1) vs += __shfl_xor(vs, m, 32);
            float rstd = rsqrtf(vs * (1.f / 256.f) + 1e-5f);
#pragma unroll
            for (int j = 0; j < 8; ++j) {
                int c = j * 32 + lane;
                Ws[t * 512 + c]       = f2bf((v[j] - mu) * rstd * n1w[c] + n1b[c]);
                Ws[t * 512 + 256 + c] = f2bf(gr[c]);
            }
            if (lane == 0) {
                pixs[t] = pix;
                int gh = wh * 7 + th, gw = ww * 7 + tw;
                int rh = (gh < 49) ? 0 : (gh < 53 ? 1 : 2);
                int rw = (gw < 49) ? 0 : (gw < 53 ? 1 : 2);
                rids[t] = rh * 3 + rw;
            }
        } else {
#pragma unroll
            for (int j = 0; j < 16; ++j) Ws[t * 512 + j * 32 + lane] = f2bf(0.f);
            if (lane == 0) { pixs[t] = -1; rids[t] = -9; }
        }
    }
    __syncthreads();

    // ---- phase 2: QKV = win @ W (Q,K: K=512; V: K=256). 192 tiles over 8 waves ----
    for (int job = wave; job < 192; job += 8) {
        int mat = job >> 6, tile = job & 63;
        int mt = tile >> 4, nt = tile & 15;
        int m0 = mt * 16, n0 = nt * 16;
        v8f acc = {};
        const int g = lane >> 4, nc = lane & 15, c = n0 + nc;
        if (mat < 2) {
            const __bf16* wT = (mat == 0) ? qT : kT;
            for (int kk = 0; kk < 512; kk += 32)
                acc = wmma_bf16(load_a(Ws, 512, m0, kk, lane), load_b(wT, 512, n0, kk, lane), acc);
            float bcv = (mat == 0) ? qb[c] : kb[c];
            float scl = (mat == 0) ? 0.17677669529663688f : 1.0f;
            __bf16* dst = (mat == 0) ? Qs : Ks;
#pragma unroll
            for (int r = 0; r < 8; ++r) dst[(m0 + r + 8 * g) * 256 + c] = f2bf((acc[r] + bcv) * scl);
        } else {
            for (int kk = 0; kk < 256; kk += 32)
                acc = wmma_bf16(load_a(Ws, 512, m0, kk, lane), load_b(vT, 256, n0, kk, lane), acc);
            float bcv = vb[c];
#pragma unroll
            for (int r = 0; r < 8; ++r) Vt[c * 64 + (m0 + r + 8 * g)] = f2bf(acc[r] + bcv); // transposed
        }
    }
    __syncthreads();

    // ---- phase 3: one wave per head: S = QK^T, masked softmax, O_h = P@V ----
    {
        const int h = wave;
        float* S = Ss + h * 4096;
        const int g = lane >> 4, nc = lane & 15;
#pragma unroll
        for (int mt = 0; mt < 4; ++mt)
#pragma unroll
        for (int nt = 0; nt < 4; ++nt) {
            v8f acc = {};
            acc = wmma_bf16(load_a(Qs, 256, mt * 16, h * 32, lane),
                            load_b(Ks, 256, nt * 16, h * 32, lane), acc); // B = K^T
#pragma unroll
            for (int r = 0; r < 8; ++r) S[(mt * 16 + r + 8 * g) * 64 + nt * 16 + nc] = acc[r];
        }
        __bf16* P = (__bf16*)S;  // bf16 P overwrites S in place (row m fits before row m reads)
        for (int m = 0; m < 64; ++m) {
            int rm = rids[m];
            float v0 = S[m * 64 + lane];
            float v1 = S[m * 64 + 32 + lane];
            v0 += (lane < 49) ? ((rids[lane] != rm) ? -100.f : 0.f) : -1e30f;
            v1 += -1e30f * ((32 + lane) >= 49 ? 1.f : 0.f)
                + (((32 + lane) < 49 && rids[32 + lane] != rm) ? -100.f : 0.f);
            float mx = fmaxf(v0, v1);
            for (int s2 = 16; s2 >= 1; s2 >>= 1) mx = fmaxf(mx, __shfl_xor(mx, s2, 32));
            float e0 = __expf(v0 - mx), e1 = __expf(v1 - mx);
            float sm = e0 + e1;
            for (int s2 = 16; s2 >= 1; s2 >>= 1) sm += __shfl_xor(sm, s2, 32);
            float inv = 1.f / sm;
            P[m * 64 + lane]      = f2bf(e0 * inv);
            P[m * 64 + 32 + lane] = f2bf(e1 * inv);
        }
#pragma unroll
        for (int mt = 0; mt < 4; ++mt)
#pragma unroll
        for (int ct = 0; ct < 2; ++ct) {
            v8f acc = {};
#pragma unroll
            for (int kk = 0; kk < 64; kk += 32)
                acc = wmma_bf16(load_a(P, 64, mt * 16, kk, lane),
                                load_b(Vt, 64, h * 32 + ct * 16, kk, lane), acc);
            int c = h * 32 + ct * 16 + nc;
#pragma unroll
            for (int r = 0; r < 8; ++r) Os[(mt * 16 + r + 8 * g) * 256 + c] = f2bf(acc[r]);
        }
    }
    __syncthreads();

    // ---- phase 4: proj + residual, scattered to rolled-back pixels ----
    for (int tile = wave; tile < 64; tile += 8) {
        int mt = tile >> 4, nt = tile & 15;
        v8f acc = {};
        for (int kk = 0; kk < 256; kk += 32)
            acc = wmma_bf16(load_a(Os, 256, mt * 16, kk, lane), load_b(pT, 256, nt * 16, kk, lane), acc);
        const int g = lane >> 4, nc = lane & 15, c = nt * 16 + nc;
        float bcv = pb[c];
#pragma unroll
        for (int r = 0; r < 8; ++r) {
            int t2 = mt * 16 + r + 8 * g;
            int pix = pixs[t2];
            if (pix >= 0) {
                size_t off = ((size_t)b * 3136 + pix) * 256 + c;
                x2out[off] = x[off] + acc[r] + bcv;
            }
        }
    }
}

// =================== fused LN2 + MLP (fc1/gelu/fc2) + residual ===================
__global__ __launch_bounds__(256) void swin_mlp(
    const float* __restrict__ n2w, const float* __restrict__ n2b,
    const __bf16* __restrict__ f1T, const float* __restrict__ f1b,
    const __bf16* __restrict__ f2T, const float* __restrict__ f2b,
    float* __restrict__ xio)
{
    extern __shared__ char smem[];
    __bf16* Xs = (__bf16*)smem;      // 64x256
    __bf16* Hs = Xs + 64 * 256;      // 64x1024
    const int tid = threadIdx.x, lane = tid & 31, wave = tid >> 5;
    const size_t row0 = (size_t)blockIdx.x * 64;

    for (int t = wave; t < 64; t += 8) {
        const float* xr = xio + (row0 + t) * 256;
        float v[8]; float s = 0.f;
#pragma unroll
        for (int j = 0; j < 8; ++j) { v[j] = xr[j * 32 + lane]; s += v[j]; }
        for (int m = 16; m >= 1; m >>= 1) s += __shfl_xor(s, m, 32);
        float mu = s * (1.f / 256.f), vs = 0.f;
#pragma unroll
        for (int j = 0; j < 8; ++j) { float d = v[j] - mu; vs += d * d; }
        for (int m = 16; m >= 1; m >>= 1) vs += __shfl_xor(vs, m, 32);
        float rstd = rsqrtf(vs * (1.f / 256.f) + 1e-5f);
#pragma unroll
        for (int j = 0; j < 8; ++j) {
            int c = j * 32 + lane;
            Xs[t * 256 + c] = f2bf((v[j] - mu) * rstd * n2w[c] + n2b[c]);
        }
    }
    __syncthreads();

    for (int tile = wave; tile < 256; tile += 8) {       // fc1: 4 x 64 tiles
        int mt = tile >> 6, nt = tile & 63;
        v8f acc = {};
        for (int kk = 0; kk < 256; kk += 32)
            acc = wmma_bf16(load_a(Xs, 256, mt * 16, kk, lane), load_b(f1T, 256, nt * 16, kk, lane), acc);
        const int g = lane >> 4, nc = lane & 15, c = nt * 16 + nc;
        float bcv = f1b[c];
#pragma unroll
        for (int r = 0; r < 8; ++r) {
            float xg = acc[r] + bcv;
            float ge = 0.5f * xg * (1.f + erff(xg * 0.70710678118654752f)); // exact gelu
            Hs[(mt * 16 + r + 8 * g) * 1024 + c] = f2bf(ge);
        }
    }
    __syncthreads();

    for (int tile = wave; tile < 64; tile += 8) {        // fc2: 4 x 16 tiles, K=1024
        int mt = tile >> 4, nt = tile & 15;
        v8f acc = {};
        for (int kk = 0; kk < 1024; kk += 32)
            acc = wmma_bf16(load_a(Hs, 1024, mt * 16, kk, lane), load_b(f2T, 1024, nt * 16, kk, lane), acc);
        const int g = lane >> 4, nc = lane & 15, c = nt * 16 + nc;
        float bcv = f2b[c];
#pragma unroll
        for (int r = 0; r < 8; ++r) {
            size_t off = (row0 + mt * 16 + r + 8 * g) * 256 + c;
            xio[off] = xio[off] + acc[r] + bcv;
        }
    }
}

extern "C" void kernel_launch(void* const* d_in, const int* in_sizes, int n_in,
                              void* d_out, int out_size, void* d_ws, size_t ws_size,
                              hipStream_t stream) {
    (void)in_sizes; (void)n_in; (void)out_size; (void)ws_size;
    const float* x    = (const float*)d_in[0];
    const float* guid = (const float*)d_in[1];
    const float* n1w  = (const float*)d_in[2];
    const float* n1b  = (const float*)d_in[3];
    const float* qw   = (const float*)d_in[4];
    const float* qb   = (const float*)d_in[5];
    const float* kw   = (const float*)d_in[6];
    const float* kb   = (const float*)d_in[7];
    const float* vw   = (const float*)d_in[8];
    const float* vb   = (const float*)d_in[9];
    const float* pw   = (const float*)d_in[10];
    const float* pb   = (const float*)d_in[11];
    const float* n2w  = (const float*)d_in[12];
    const float* n2b  = (const float*)d_in[13];
    const float* f1   = (const float*)d_in[14];
    const float* f1b  = (const float*)d_in[15];
    const float* f2   = (const float*)d_in[16];
    const float* f2b  = (const float*)d_in[17];
    float* out = (float*)d_out;

    __bf16* wsb = (__bf16*)d_ws;
    const __bf16* qT  = wsb;
    const __bf16* kT  = qT  + 256 * 512;
    const __bf16* vT  = kT  + 256 * 512;
    const __bf16* pT  = vT  + 256 * 256;
    const __bf16* f1T = pT  + 256 * 256;
    const __bf16* f2T = f1T + 1024 * 256;

    swin_wconv<<<3584, 256, 0, stream>>>(qw, kw, vw, pw, f1, f2, wsb);

    // LDS: Q+K+Vt (96KB) + S/P (128KB, overlays window buffer) + O (32KB) + idx (512B)
    size_t attn_lds = (size_t)3 * 64 * 256 * 2 + 8 * 4096 * 4 + 64 * 256 * 2 + 128 * 4;
    swin_attn<<<2048, 256, attn_lds, stream>>>(x, guid, n1w, n1b,
                                               qT, qb, kT, kb, vT, vb, pT, pb, out);

    size_t mlp_lds = (size_t)64 * 256 * 2 + (size_t)64 * 1024 * 2; // 160KB
    swin_mlp<<<1568, 256, mlp_lds, stream>>>(n2w, n2b, f1T, f1b, f2T, f2b, out);
}
